// GraphConvLayer_41360535061191
// MI455X (gfx1250) — matrix operand order repaired and checked by reading, not compiled, not run on hardware
//
#include <hip/hip_runtime.h>

typedef __attribute__((ext_vector_type(16))) _Float16 v16h;
typedef __attribute__((ext_vector_type(8)))  _Float16 v8h;
typedef __attribute__((ext_vector_type(8)))  float    v8f;

#define WMMA16(a,b,c) __builtin_amdgcn_wmma_f32_16x16x32_f16(false,(a),false,(b),(short)0,(c),false,false)

// ---------------------------------------------------------------------------
// Pack fp32 row-major weight [K, NC] into f16 B-fragment order:
// out[((kt*ntiles + nt)*32 + lane)*16 + h] = w[(kt*32 + (h>>3)*16 + (lane>>4)*8 + (h&7))*NC + nt*16 + (lane&15)]
// so a WMMA B fragment is one contiguous 32B load per lane.
// ---------------------------------------------------------------------------
__global__ void pack_w(const float* __restrict__ w, _Float16* __restrict__ o,
                       int NC, int ktiles, int ntiles) {
  int t = blockIdx.x * 256 + threadIdx.x;
  if (t >= ktiles * ntiles * 512) return;
  int h  = t & 15;
  int l  = (t >> 4) & 31;
  int fr = t >> 9;
  int nt = fr % ntiles;
  int kt = fr / ntiles;
  int k = kt * 32 + (h >> 3) * 16 + (l >> 4) * 8 + (h & 7);
  int c = nt * 16 + (l & 15);
  o[t] = (_Float16)w[(long long)k * NC + c];
}

__global__ void zero_f32(float* p, long long count) {
  long long t = (long long)blockIdx.x * 256 + threadIdx.x;
  if (t < count) p[t] = 0.f;
}

// ---------------------------------------------------------------------------
// Edge kernel: per wave, 16 edges.
// GEMM1 [16,192]x[192,128] -> LN -> SiLU -> GEMM2 [16,128]x[128,64]
// + edge_attr residual; store edge_new; atomic scatter-add into agg[dst].
// ---------------------------------------------------------------------------
__global__ __launch_bounds__(256) void edge_kernel(
    const float* __restrict__ x, const long long* __restrict__ eidx,
    const float* __restrict__ ea,
    const _Float16* __restrict__ w1, const float* __restrict__ b1,
    const float* __restrict__ g,  const float* __restrict__ bt,
    const _Float16* __restrict__ w2, const float* __restrict__ b2,
    float* agg, float* __restrict__ eout, long long E)
{
  __shared__ __align__(16) _Float16 lds_h[8][16 * 128];
  const int lane = threadIdx.x & 31;
  const int wave = threadIdx.x >> 5;
  const int n  = lane & 15;
  const int hi = lane >> 4;
  const long long e0 = ((long long)blockIdx.x * 8 + wave) * 16;
  if (e0 >= E) return;
  const long long e   = e0 + n;        // gather row owned by this lane
  const long long src = eidx[e];
  const long long dst = eidx[E + e];
  const int dsti = (int)dst;

  // ---- GEMM1: acc[nt] = edge_in @ w1 + b1 -------------------------------
  v8f acc[8];
  #pragma unroll
  for (int nt = 0; nt < 8; ++nt) {
    float bb = b1[nt * 16 + n];
    #pragma unroll
    for (int r = 0; r < 8; ++r) acc[nt][r] = bb;
  }
  #pragma unroll
  for (int ki = 0; ki < 6; ++ki) {
    v16h a;
    #pragma unroll
    for (int ch = 0; ch < 2; ++ch) {
      const int base = ki * 32 + ch * 16;        // compile-time per unroll site
      const float* p;
      if (base < 64)       p = x  + src * 64 + base +        hi * 8;
      else if (base < 128) p = x  + dst * 64 + (base - 64) + hi * 8;
      else                 p = ea + e   * 64 + (base - 128) + hi * 8;
      float4 f0 = *(const float4*)p;
      float4 f1 = *(const float4*)(p + 4);
      const int o = ch * 8;
      a[o+0]=(_Float16)f0.x; a[o+1]=(_Float16)f0.y; a[o+2]=(_Float16)f0.z; a[o+3]=(_Float16)f0.w;
      a[o+4]=(_Float16)f1.x; a[o+5]=(_Float16)f1.y; a[o+6]=(_Float16)f1.z; a[o+7]=(_Float16)f1.w;
    }
    #pragma unroll
    for (int nt = 0; nt < 8; ++nt) {
      v16h b = *(const v16h*)(w1 + ((ki * 8 + nt) * 32 + lane) * 16);
      acc[nt] = WMMA16(a, b, acc[nt]);
    }
  }

  // ---- LayerNorm over 128 (rows split: lanes 0-15 hold cols of rows hi*8+r)
  float mu[8], rs[8];
  #pragma unroll
  for (int r = 0; r < 8; ++r) {
    float s = 0.f, s2 = 0.f;
    #pragma unroll
    for (int nt = 0; nt < 8; ++nt) { float v = acc[nt][r]; s += v; s2 += v * v; }
    #pragma unroll
    for (int off = 1; off < 16; off <<= 1) {
      s  += __shfl_xor(s,  off, 16);
      s2 += __shfl_xor(s2, off, 16);
    }
    float m_  = s  * (1.f / 128.f);
    float var = s2 * (1.f / 128.f) - m_ * m_;
    mu[r] = m_;
    rs[r] = rsqrtf(var + 1e-5f);
  }

  // ---- scale/shift + SiLU, stage to LDS in A-row-major [row][col] -------
  _Float16* hrow = &lds_h[wave][0];
  #pragma unroll
  for (int nt = 0; nt < 8; ++nt) {
    float gg = g[nt * 16 + n], bb = bt[nt * 16 + n];
    #pragma unroll
    for (int r = 0; r < 8; ++r) {
      float v = (acc[nt][r] - mu[r]) * rs[r] * gg + bb;
      v = v / (1.f + __expf(-v));
      hrow[(hi * 8 + r) * 128 + nt * 16 + n] = (_Float16)v;
    }
  }

  // ---- GEMM2: acc2[nt] = h @ w2 + b2 ------------------------------------
  v8f acc2[4];
  #pragma unroll
  for (int nt = 0; nt < 4; ++nt) {
    float bb = b2[nt * 16 + n];
    #pragma unroll
    for (int r = 0; r < 8; ++r) acc2[nt][r] = bb;
  }
  const _Float16* hp = &lds_h[wave][n * 128];   // A row m = n
  #pragma unroll
  for (int ki = 0; ki < 4; ++ki) {
    v8h c0 = *(const v8h*)(hp + ki * 32 +      hi * 8);
    v8h c1 = *(const v8h*)(hp + ki * 32 + 16 + hi * 8);
    v16h a;
    #pragma unroll
    for (int i = 0; i < 8; ++i) { a[i] = c0[i]; a[i + 8] = c1[i]; }
    #pragma unroll
    for (int nt = 0; nt < 4; ++nt) {
      v16h b = *(const v16h*)(w2 + ((ki * 4 + nt) * 32 + lane) * 16);
      acc2[nt] = WMMA16(a, b, acc2[nt]);
    }
  }

  // ---- residual, store edge_new, scatter-add into agg -------------------
  #pragma unroll
  for (int nt = 0; nt < 4; ++nt) {
    const int c = nt * 16 + n;
    #pragma unroll
    for (int r = 0; r < 8; ++r) {
      const long long row = e0 + hi * 8 + r;
      float v = acc2[nt][r] + ea[row * 64 + c];
      eout[row * 64 + c] = v;
      int drow = __shfl(dsti, hi * 8 + r, 16);
      atomicAdd(agg + (long long)drow * 64 + c, v);
    }
  }
}

// ---------------------------------------------------------------------------
// Node kernel: per wave, 16 nodes.
// GEMM1 [16,128]x[128,128] on {x,agg} -> LN -> SiLU -> GEMM2 [16,128]x[128,64]
// + x residual. agg aliases xout (read-before-write per tile).
// ---------------------------------------------------------------------------
__global__ __launch_bounds__(256) void node_kernel(
    const float* __restrict__ x, const float* agg,
    const _Float16* __restrict__ w1, const float* __restrict__ b1,
    const float* __restrict__ g,  const float* __restrict__ bt,
    const _Float16* __restrict__ w2, const float* __restrict__ b2,
    float* xout, long long N)
{
  __shared__ __align__(16) _Float16 lds_h[8][16 * 128];
  const int lane = threadIdx.x & 31;
  const int wave = threadIdx.x >> 5;
  const int n  = lane & 15;
  const int hi = lane >> 4;
  const long long i0 = ((long long)blockIdx.x * 8 + wave) * 16;
  if (i0 >= N) return;
  const long long i = i0 + n;

  v8f acc[8];
  #pragma unroll
  for (int nt = 0; nt < 8; ++nt) {
    float bb = b1[nt * 16 + n];
    #pragma unroll
    for (int r = 0; r < 8; ++r) acc[nt][r] = bb;
  }
  #pragma unroll
  for (int ki = 0; ki < 4; ++ki) {
    v16h a;
    #pragma unroll
    for (int ch = 0; ch < 2; ++ch) {
      const int base = ki * 32 + ch * 16;
      const float* p = (base < 64) ? (x   + i * 64 + base +        hi * 8)
                                   : (agg + i * 64 + (base - 64) + hi * 8);
      float4 f0 = *(const float4*)p;
      float4 f1 = *(const float4*)(p + 4);
      const int o = ch * 8;
      a[o+0]=(_Float16)f0.x; a[o+1]=(_Float16)f0.y; a[o+2]=(_Float16)f0.z; a[o+3]=(_Float16)f0.w;
      a[o+4]=(_Float16)f1.x; a[o+5]=(_Float16)f1.y; a[o+6]=(_Float16)f1.z; a[o+7]=(_Float16)f1.w;
    }
    #pragma unroll
    for (int nt = 0; nt < 8; ++nt) {
      v16h b = *(const v16h*)(w1 + ((ki * 8 + nt) * 32 + lane) * 16);
      acc[nt] = WMMA16(a, b, acc[nt]);
    }
  }

  float mu[8], rs[8];
  #pragma unroll
  for (int r = 0; r < 8; ++r) {
    float s = 0.f, s2 = 0.f;
    #pragma unroll
    for (int nt = 0; nt < 8; ++nt) { float v = acc[nt][r]; s += v; s2 += v * v; }
    #pragma unroll
    for (int off = 1; off < 16; off <<= 1) {
      s  += __shfl_xor(s,  off, 16);
      s2 += __shfl_xor(s2, off, 16);
    }
    float m_  = s  * (1.f / 128.f);
    float var = s2 * (1.f / 128.f) - m_ * m_;
    mu[r] = m_;
    rs[r] = rsqrtf(var + 1e-5f);
  }

  _Float16* hrow = &lds_h[wave][0];
  #pragma unroll
  for (int nt = 0; nt < 8; ++nt) {
    float gg = g[nt * 16 + n], bb = bt[nt * 16 + n];
    #pragma unroll
    for (int r = 0; r < 8; ++r) {
      float v = (acc[nt][r] - mu[r]) * rs[r] * gg + bb;
      v = v / (1.f + __expf(-v));
      hrow[(hi * 8 + r) * 128 + nt * 16 + n] = (_Float16)v;
    }
  }

  v8f acc2[4];
  #pragma unroll
  for (int nt = 0; nt < 4; ++nt) {
    float bb = b2[nt * 16 + n];
    #pragma unroll
    for (int r = 0; r < 8; ++r) acc2[nt][r] = bb;
  }
  const _Float16* hp = &lds_h[wave][n * 128];
  #pragma unroll
  for (int ki = 0; ki < 4; ++ki) {
    v8h c0 = *(const v8h*)(hp + ki * 32 +      hi * 8);
    v8h c1 = *(const v8h*)(hp + ki * 32 + 16 + hi * 8);
    v16h a;
    #pragma unroll
    for (int i2 = 0; i2 < 8; ++i2) { a[i2] = c0[i2]; a[i2 + 8] = c1[i2]; }
    #pragma unroll
    for (int nt = 0; nt < 4; ++nt) {
      v16h b = *(const v16h*)(w2 + ((ki * 4 + nt) * 32 + lane) * 16);
      acc2[nt] = WMMA16(a, b, acc2[nt]);
    }
  }

  #pragma unroll
  for (int nt = 0; nt < 4; ++nt) {
    const int c = nt * 16 + n;
    #pragma unroll
    for (int r = 0; r < 8; ++r) {
      const long long row = i0 + hi * 8 + r;
      xout[row * 64 + c] = acc2[nt][r] + x[row * 64 + c];
    }
  }
}

// ---------------------------------------------------------------------------
extern "C" void kernel_launch(void* const* d_in, const int* in_sizes, int n_in,
                              void* d_out, int out_size, void* d_ws, size_t ws_size,
                              hipStream_t stream) {
  const float*     x    = (const float*)d_in[0];
  const long long* eidx = (const long long*)d_in[1];
  const float*     ea   = (const float*)d_in[2];
  const float*     ew1  = (const float*)d_in[3];
  const float*     eb1  = (const float*)d_in[4];
  const float*     eg   = (const float*)d_in[5];
  const float*     ebt  = (const float*)d_in[6];
  const float*     ew2  = (const float*)d_in[7];
  const float*     eb2  = (const float*)d_in[8];
  const float*     nw1  = (const float*)d_in[9];
  const float*     nb1  = (const float*)d_in[10];
  const float*     ng   = (const float*)d_in[11];
  const float*     nbt  = (const float*)d_in[12];
  const float*     nw2  = (const float*)d_in[13];
  const float*     nb2  = (const float*)d_in[14];

  const long long N = in_sizes[0] / 64;   // 50000
  const long long E = in_sizes[1] / 2;    // 800000

  // ws: packed f16 weights (112 KB total)
  _Float16* ws   = (_Float16*)d_ws;
  _Float16* ew1h = ws;              // 6*8*512 = 24576 halves
  _Float16* ew2h = ws + 24576;      // 4*4*512 =  8192
  _Float16* nw1h = ws + 32768;      // 4*8*512 = 16384
  _Float16* nw2h = ws + 49152;      //           8192

  float* xnew = (float*)d_out;            // [N,64]; doubles as agg accumulator
  float* enew = (float*)d_out + N * 64;   // [E,64]

  pack_w<<<(24576 + 255) / 256, 256, 0, stream>>>(ew1, ew1h, 128, 6, 8);
  pack_w<<<( 8192 + 255) / 256, 256, 0, stream>>>(ew2, ew2h,  64, 4, 4);
  pack_w<<<(16384 + 255) / 256, 256, 0, stream>>>(nw1, nw1h, 128, 4, 8);
  pack_w<<<( 8192 + 255) / 256, 256, 0, stream>>>(nw2, nw2h,  64, 4, 4);

  zero_f32<<<(int)((N * 64 + 255) / 256), 256, 0, stream>>>(xnew, N * 64);

  edge_kernel<<<(int)((E + 127) / 128), 256, 0, stream>>>(
      x, eidx, ea, ew1h, eb1, eg, ebt, ew2h, eb2, xnew, enew, E);

  node_kernel<<<(int)((N + 127) / 128), 256, 0, stream>>>(
      x, xnew, nw1h, nb1, ng, nbt, nw2h, nb2, xnew, N);
}
